// LinearTemporalSelfAttention_781684048568
// MI455X (gfx1250) — compile-verified, matrix operand order checked by reading, not run on hardware
//
#include <hip/hip_runtime.h>
#include <hip/hip_bf16.h>

typedef __bf16 bf16;
typedef __attribute__((ext_vector_type(16))) __bf16 v16bf;
typedef __attribute__((ext_vector_type(8)))  float   v8f;
typedef __attribute__((ext_vector_type(2)))  float   v2f;

#define CB 4
#define CT 8192
#define CD 512
#define CH 8
#define CTE 2048
#define CM (CB*CT)
#define NEG_BIG (-3.0e38f)
#define LDSROW 72     // 64 bf16 payload + 8 bf16 (16B) TDM pad -> conflict-free ds_load_b128

#ifndef __has_builtin
#define __has_builtin(x) 0
#endif
#if defined(__AMDGCN__) && __has_builtin(__builtin_amdgcn_tensor_load_to_lds)
#define HAVE_TDM 1
#else
#define HAVE_TDM 0
#endif

// ---------- WMMA fragment helpers ----------
union FragBF { v16bf v; int4 q[2]; };

// global-memory fragment: lanep already includes (lane&15)*ld + (lane>>4)*8
__device__ __forceinline__ v16bf load_frag_at(const bf16* lanep) {
  FragBF f;
  f.q[0] = *(const int4*)(lanep);
  f.q[1] = *(const int4*)(lanep + 16);
  return f.v;
}

// LDS fragment: index directly into the __shared__ array so clang keeps
// addrspace(3) and emits ds_load_b128 (no flat_load, no loadcnt coupling).
#define FRAG_FROM_LDS(dst, arr, idx)                         \
  do { FragBF _f;                                            \
       _f.q[0] = *(const int4*)&(arr)[(idx)];                \
       _f.q[1] = *(const int4*)&(arr)[(idx) + 16];           \
       (dst) = _f.v; } while (0)

__device__ __forceinline__ v8f wmma_bf16(v16bf a, v16bf b, v8f c) {
  return __builtin_amdgcn_wmma_f32_16x16x32_bf16(false, a, false, b, (short)0, c, false, false);
}

// ---------- Tensor Data Mover: 3D tile (k contiguous, n rows, mat planes) -> LDS ----------
#if HAVE_TDM
typedef __attribute__((ext_vector_type(4))) unsigned int v4u_t;
typedef __attribute__((ext_vector_type(8))) int v8i_t;
typedef __attribute__((ext_vector_type(4))) int v4i_t;

__device__ __forceinline__ void tdm_load_panel(unsigned lds_off, const bf16* gsrc,
                                               unsigned tdim0, unsigned tdim1, unsigned nmats,
                                               unsigned long long stride0,
                                               unsigned long long stride1) {
  unsigned long long ga = (unsigned long long)(const void*)gsrc;
  v4u_t g0;
  g0.x = 1u;                                              // count=1 (valid user D#)
  g0.y = lds_off;                                         // LDS byte address
  g0.z = (unsigned)ga;                                    // global_addr[31:0]
  g0.w = (unsigned)((ga >> 32) & 0x1ffffffull) | (2u << 30);  // addr[56:32] | type=2
  v8i_t g1;
  g1[0] = (int)((1u << 16) | (1u << 20) | (4u << 22) | (3u << 25));
          // data_size=2B | pad_enable | pad_interval=32 DW (128B) | pad_amount=4 DW (16B)
  g1[1] = (int)((tdim0 & 0xffffu) << 16);                 // tensor_dim0[15:0]
  g1[2] = (int)((tdim0 >> 16) | ((tdim1 & 0xffffu) << 16));
  g1[3] = (int)((tdim1 >> 16) | (64u << 16));             // tile_dim0 = 64 (k)
  g1[4] = (int)(64u | (nmats << 16));                     // tile_dim1=64 (n), tile_dim2=mats
  g1[5] = (int)(stride0 & 0xffffffffull);                 // tensor_dim0_stride[31:0]
  g1[6] = (int)(((stride0 >> 32) & 0xffffull) | ((stride1 & 0xffffull) << 16));
  g1[7] = (int)((stride1 >> 16) & 0xffffffffull);         // tensor_dim1_stride[47:16]
  v4i_t g2;
  g2[0] = (int)nmats;                                     // tensor_dim2
  g2[1] = 0;
  g2[2] = 0;
  g2[3] = 0;
  v4i_t g3 = {0, 0, 0, 0};
#if __clang_major__ >= 23
  v8i_t g4 = {0, 0, 0, 0, 0, 0, 0, 0};
  __builtin_amdgcn_tensor_load_to_lds(g0, g1, g2, g3, g4, 0);
#else
  __builtin_amdgcn_tensor_load_to_lds(g0, g1, g2, g3, 0);
#endif
}
#endif

__device__ __forceinline__ unsigned lds_off_of(const void* p) {
  return (unsigned)(unsigned long long)p;   // flat LDS addr: low 32 bits = LDS offset
}

// Stage nmats planes of a 64(n) x 64(k) bf16 tile into LDS (row stride LDSROW).
__device__ __forceinline__ void stage_panel(bf16* dst, const bf16* src, int nmats,
                                            unsigned tdim0, unsigned tdim1,
                                            size_t rowstride, size_t matstride) {
#if HAVE_TDM
  if (threadIdx.x < 32)
    tdm_load_panel(lds_off_of(dst), src, tdim0, tdim1, (unsigned)nmats,
                   (unsigned long long)rowstride, (unsigned long long)matstride);
#else
  for (int m = 0; m < nmats; ++m)
    for (int idx = threadIdx.x; idx < 64 * 8; idx += blockDim.x) {
      int r = idx >> 3, c = idx & 7;
      *(int4*)(dst + (size_t)(m * 64 + r) * LDSROW + c * 8) =
          *(const int4*)(src + (size_t)m * matstride + (size_t)r * rowstride + c * 8);
    }
#endif
}

__device__ __forceinline__ void wait_panel_le1(void) {
#if HAVE_TDM
  __builtin_amdgcn_s_wait_tensorcnt(1);
#endif
  __syncthreads();
}
__device__ __forceinline__ void wait_panel_le0(void) {
#if HAVE_TDM
  __builtin_amdgcn_s_wait_tensorcnt(0);
#endif
  __syncthreads();
}

// ---------- block reduction ----------
__device__ __forceinline__ float block_reduce_sum(float v, float* smem) {
  int tid = threadIdx.x;
  #pragma unroll
  for (int m = 16; m >= 1; m >>= 1) v += __shfl_xor(v, m, 32);
  if ((tid & 31) == 0) smem[tid >> 5] = v;
  __syncthreads();
  int nw = blockDim.x >> 5;
  v = (tid < nw) ? smem[tid] : 0.0f;
  if (tid < 32) {
    #pragma unroll
    for (int m = 16; m >= 1; m >>= 1) v += __shfl_xor(v, m, 32);
  }
  if (tid == 0) smem[0] = v;
  __syncthreads();
  v = smem[0];
  __syncthreads();
  return v;
}

// ---------- 1. weight convert ----------
__global__ void wconv_kernel(const float* wq, const float* wk, const float* wv,
                             const float* wo, bf16* wt) {
  int id = blockIdx.x * 256 + threadIdx.x;
  int mat = id >> 18;
  int rem = id & 262143;
  int n = rem >> 9, k = rem & 511;
  const float* W = (mat == 0) ? wq : (mat == 1) ? wk : (mat == 2) ? wv : wo;
  wt[(size_t)mat * 262144 + (size_t)n * 512 + k] = (bf16)W[(size_t)k * 512 + n];
}

// ---------- 2. LayerNorm(x) -> xn bf16 ----------
__global__ void ln1_kernel(const float* __restrict__ x, const float* g,
                           const float* be, bf16* xn) {
  __shared__ float sm[32];
  size_t row = blockIdx.x;
  const float* xr = x + row * CD;
  int tid = threadIdx.x;
  float v0 = xr[tid], v1 = xr[tid + 256];
  float mu = block_reduce_sum(v0 + v1, sm) * (1.0f / CD);
  float d0 = v0 - mu, d1 = v1 - mu;
  float var = block_reduce_sum(d0 * d0 + d1 * d1, sm) * (1.0f / CD);
  float rs = rsqrtf(var + 1e-5f);
  xn[row * CD + tid]       = (bf16)(d0 * rs * g[tid] + be[tid]);
  xn[row * CD + tid + 256] = (bf16)(d1 * rs * g[tid + 256] + be[tid + 256]);
}

// ---------- 3. fused QKV GEMM: TDM-staged weights, bf16 WMMA, fused epilogue ----------
__global__ void __launch_bounds__(256) qkv_gemm_kernel(
    const bf16* __restrict__ xn, const bf16* __restrict__ wt3,
    const float* bq, const float* bk, const float* bv,
    const float* __restrict__ mask, bf16* qb, float* kf, float* vf) {
  __shared__ __align__(16) bf16 wbuf[2][3 * 64 * LDSROW];   // 54 KB double-buffered
  int n0 = blockIdx.x * 64;                                 // head h = blockIdx.x
  int wave = threadIdx.x >> 5, lane = threadIdx.x & 31;
  int cl = lane & 15, lg = lane >> 4;
  int m0 = blockIdx.y * 128 + wave * 16;
  v8f aq[4] = {}, akk[4] = {}, av[4] = {};
  const bf16* aLane = xn + (size_t)(m0 + cl) * CD + lg * 8;
  const unsigned lbase = (unsigned)(cl * LDSROW + lg * 8);  // lane part of LDS index
  const bf16* wsrc = wt3 + (size_t)n0 * CD;

  stage_panel(&wbuf[0][0], wsrc, 3, 512, 512, 512, 262144);
  for (int kp = 0; kp < 8; ++kp) {
    if (kp < 7)
      stage_panel(&wbuf[(kp + 1) & 1][0], wsrc + (kp + 1) * 64, 3, 512, 512, 512, 262144);
    if (kp < 7) wait_panel_le1(); else wait_panel_le0();
    int buf = kp & 1;
    __builtin_prefetch(aLane + kp * 64 + 256, 0, 1);
    // hoist both A fragments of this K-panel so global loads overlap the LDS-fed WMMAs
    v16bf a0 = load_frag_at(aLane + kp * 64);
    v16bf a1 = load_frag_at(aLane + kp * 64 + 32);
    #pragma unroll
    for (int ks = 0; ks < 2; ++ks) {
      v16bf a = ks ? a1 : a0;
      #pragma unroll
      for (int t = 0; t < 4; ++t) {
        unsigned idx = lbase + (unsigned)((t * 16) * LDSROW + ks * 32);
        v16bf b0, b1, b2;
        FRAG_FROM_LDS(b0, wbuf[buf], idx);
        FRAG_FROM_LDS(b1, wbuf[buf], idx + 64u * LDSROW);
        FRAG_FROM_LDS(b2, wbuf[buf], idx + 128u * LDSROW);
        aq[t]  = wmma_bf16(a, b0, aq[t]);
        akk[t] = wmma_bf16(a, b1, akk[t]);
        av[t]  = wmma_bf16(a, b2, av[t]);
      }
    }
    __syncthreads();                 // reads done before TDM reuses this buffer
  }

  #pragma unroll
  for (int r = 0; r < 8; ++r) {
    int row = m0 + r + 8 * lg;
    float mk = mask[row];
    float qv[4]; float mx = NEG_BIG;
    #pragma unroll
    for (int t = 0; t < 4; ++t) {
      qv[t] = aq[t][r] + bq[n0 + t * 16 + cl];
      mx = fmaxf(mx, qv[t]);
    }
    #pragma unroll
    for (int s = 8; s >= 1; s >>= 1) mx = fmaxf(mx, __shfl_xor(mx, s, 32));
    float sum = 0.f;
    #pragma unroll
    for (int t = 0; t < 4; ++t) { qv[t] = __expf(qv[t] - mx); sum += qv[t]; }
    #pragma unroll
    for (int s = 8; s >= 1; s >>= 1) sum += __shfl_xor(sum, s, 32);
    float inv = 1.f / sum;
    #pragma unroll
    for (int t = 0; t < 4; ++t) {
      int n = n0 + t * 16 + cl;
      size_t o = (size_t)row * CD + n;
      qb[o] = (bf16)(qv[t] * inv);
      kf[o] = akk[t][r] + bk[n] + (1.f - mk) * -1000000.0f;
      vf[o] = (av[t][r] + bv[n]) * mk;
    }
  }
}

// ---------- 4-6. softmax over time for k ----------
__global__ void ksm_partial_kernel(const float* __restrict__ kf, float* pm, float* ps) {
  int bx = blockIdx.x, tsp = blockIdx.y;
  int b = bx >> 2, dc = bx & 3;
  int d = dc * 128 + threadIdx.x;
  const float* base = kf + (size_t)b * CT * CD + d;
  float m = NEG_BIG, s = 0.f;
  int t0 = tsp * 512;
  for (int t = 0; t < 512; ++t) {
    float x = base[(size_t)(t0 + t) * CD];
    float nm = fmaxf(m, x);
    s = s * __expf(m - nm) + __expf(x - nm);
    m = nm;
  }
  int idx = (bx * 16 + tsp) * 128 + threadIdx.x;
  pm[idx] = m; ps[idx] = s;
}

__global__ void ksm_combine_kernel(const float* pm, const float* ps, float* fm, float* fs) {
  int bx = blockIdx.x, tid = threadIdx.x;
  float M = NEG_BIG;
  for (int s = 0; s < 16; ++s) M = fmaxf(M, pm[(bx * 16 + s) * 128 + tid]);
  float S = 0.f;
  for (int s = 0; s < 16; ++s)
    S += ps[(bx * 16 + s) * 128 + tid] * __expf(pm[(bx * 16 + s) * 128 + tid] - M);
  fm[bx * 128 + tid] = M; fs[bx * 128 + tid] = S;
}

__global__ void ksm_norm_kernel(float* kf, const float* fm, const float* fs) {
  int bx = blockIdx.x, tsp = blockIdx.y;
  int b = bx >> 2, dc = bx & 3;
  int d = dc * 128 + threadIdx.x;
  float M = fm[bx * 128 + threadIdx.x];
  float Si = 1.f / fs[bx * 128 + threadIdx.x];
  float* base = kf + (size_t)b * CT * CD + d;
  int t0 = tsp * 512;
  for (int t = 0; t < 512; ++t) {
    size_t off = (size_t)(t0 + t) * CD;
    base[off] = __expf(base[off] - M) * Si;
  }
}

// ---------- 7. attn = k^T v per (b,h): 64x64, K=T, f32 WMMA 16x16x4 ----------
__global__ void __launch_bounds__(256) attn_gemm_kernel(
    const float* __restrict__ kf, const float* __restrict__ vf, bf16* attnT) {
  int bh = blockIdx.x; int b = bh >> 3, h = bh & 7;
  int wave = threadIdx.x >> 5, lane = threadIdx.x & 31;
  int cl = lane & 15, lg = lane >> 4;
  int dt = wave >> 1;
  int lt0 = (wave & 1) * 2;
  const float* kp = kf + (size_t)b * CT * CD + h * 64;
  const float* vp = vf + (size_t)b * CT * CD + h * 64;
  v8f acc0 = {}, acc1 = {};
  for (int t0 = 0; t0 < CT; t0 += 4) {
    size_t r0 = (size_t)(t0 + lg * 2) * CD;
    size_t r1 = r0 + CD;
    v2f a;  a.x  = kp[r0 + dt * 16 + cl];        a.y  = kp[r1 + dt * 16 + cl];
    v2f b0; b0.x = vp[r0 + lt0 * 16 + cl];       b0.y = vp[r1 + lt0 * 16 + cl];
    v2f b1; b1.x = vp[r0 + (lt0 + 1) * 16 + cl]; b1.y = vp[r1 + (lt0 + 1) * 16 + cl];
    acc0 = __builtin_amdgcn_wmma_f32_16x16x4_f32(false, a, false, b0, (short)0, acc0, false, false);
    acc1 = __builtin_amdgcn_wmma_f32_16x16x4_f32(false, a, false, b1, (short)0, acc1, false, false);
  }
  #pragma unroll
  for (int r = 0; r < 8; ++r) {
    int d = dt * 16 + r + 8 * lg;
    int l0 = lt0 * 16 + cl, l1 = (lt0 + 1) * 16 + cl;
    attnT[((size_t)bh * 64 + l0) * 64 + d] = (bf16)acc0[r];
    attnT[((size_t)bh * 64 + l1) * 64 + d] = (bf16)acc1[r];
  }
}

// ---------- 8. y = q @ attn (bf16 WMMA, K=64; attnT TDM-staged in LDS) ----------
__global__ void __launch_bounds__(256) y_gemm_kernel(
    const bf16* __restrict__ qb, const bf16* __restrict__ attnT, float* yf) {
  __shared__ __align__(16) bf16 abuf[64 * LDSROW];          // 9 KB
  int bh = blockIdx.x; int b = bh >> 3, h = bh & 7;
  int wave = threadIdx.x >> 5, lane = threadIdx.x & 31;
  int cl = lane & 15, lg = lane >> 4;
  int m0 = blockIdx.y * 128 + wave * 16;
  v8f acc[4] = {};
  stage_panel(abuf, attnT + (size_t)bh * 64 * 64, 1, 64, 64, 64, 0);
  const bf16* aLane = qb + (size_t)(b * CT + m0 + cl) * CD + h * 64 + lg * 8;
  const unsigned lbase = (unsigned)(cl * LDSROW + lg * 8);
  wait_panel_le0();
  v16bf a0 = load_frag_at(aLane);
  v16bf a1 = load_frag_at(aLane + 32);
  #pragma unroll
  for (int ks = 0; ks < 2; ++ks) {
    v16bf a = ks ? a1 : a0;
    #pragma unroll
    for (int t = 0; t < 4; ++t) {
      v16bf bfrag;
      FRAG_FROM_LDS(bfrag, abuf, lbase + (unsigned)((t * 16) * LDSROW + ks * 32));
      acc[t] = wmma_bf16(a, bfrag, acc[t]);
    }
  }
  #pragma unroll
  for (int r = 0; r < 8; ++r) {
    size_t row = (size_t)(b * CT + m0 + r + 8 * lg);
    #pragma unroll
    for (int t = 0; t < 4; ++t)
      yf[row * CD + h * 64 + t * 16 + cl] = acc[t][r];
  }
}

// ---------- 9. emb_out = silu(emb) @ emb_W + emb_b ----------
__global__ void emb_kernel(const float* __restrict__ emb, const float* __restrict__ W,
                           const float* bb, float* out) {
  __shared__ float se[CTE];
  int b = blockIdx.x;
  int n = blockIdx.y * 256 + threadIdx.x;
  for (int e = threadIdx.x; e < CTE; e += 256) {
    float v = emb[(size_t)b * CTE + e];
    se[e] = v / (1.f + __expf(-v));
  }
  __syncthreads();
  float s = 0.f;
  for (int e = 0; e < CTE; ++e) s += se[e] * W[(size_t)e * 1024 + n];
  out[b * 1024 + n] = s + bb[n];
}

// ---------- 10. h = silu(LN(y)*(1+scale)+shift) -> bf16 ----------
__global__ void ln2mod_kernel(const float* __restrict__ y, const float* g2,
                              const float* b2, const float* __restrict__ embo,
                              bf16* hb) {
  __shared__ float sm[32];
  size_t row = blockIdx.x;
  int b = (int)(row >> 13);
  const float* yr = y + row * CD;
  int tid = threadIdx.x;
  float v0 = yr[tid], v1 = yr[tid + 256];
  float mu = block_reduce_sum(v0 + v1, sm) * (1.0f / CD);
  float d0 = v0 - mu, d1 = v1 - mu;
  float var = block_reduce_sum(d0 * d0 + d1 * d1, sm) * (1.0f / CD);
  float rs = rsqrtf(var + 1e-5f);
  #pragma unroll
  for (int i = 0; i < 2; ++i) {
    int n = tid + i * 256;
    float dv = i ? d1 : d0;
    float nv = dv * rs * g2[n] + b2[n];
    float sc = embo[b * 1024 + n], sh = embo[b * 1024 + 512 + n];
    float h = nv * (1.f + sc) + sh;
    hb[row * CD + n] = (bf16)(h / (1.f + __expf(-h)));
  }
}

// ---------- 11. out = x + silu_h @ out_W + out_b (TDM-staged, bf16 WMMA) ----------
__global__ void __launch_bounds__(256) out_gemm_kernel(
    const bf16* __restrict__ hb, const bf16* __restrict__ wot,
    const float* ob, const float* __restrict__ x, float* out) {
  __shared__ __align__(16) bf16 wbuf[2][64 * LDSROW];       // 18 KB
  int n0 = blockIdx.x * 64;
  int wave = threadIdx.x >> 5, lane = threadIdx.x & 31;
  int cl = lane & 15, lg = lane >> 4;
  int m0 = blockIdx.y * 128 + wave * 16;
  v8f acc[4] = {};
  const bf16* aLane = hb + (size_t)(m0 + cl) * CD + lg * 8;
  const unsigned lbase = (unsigned)(cl * LDSROW + lg * 8);
  const bf16* wsrc = wot + (size_t)n0 * CD;

  stage_panel(&wbuf[0][0], wsrc, 1, 512, 512, 512, 0);
  for (int kp = 0; kp < 8; ++kp) {
    if (kp < 7)
      stage_panel(&wbuf[(kp + 1) & 1][0], wsrc + (kp + 1) * 64, 1, 512, 512, 512, 0);
    if (kp < 7) wait_panel_le1(); else wait_panel_le0();
    int buf = kp & 1;
    __builtin_prefetch(aLane + kp * 64 + 256, 0, 1);
    v16bf a0 = load_frag_at(aLane + kp * 64);
    v16bf a1 = load_frag_at(aLane + kp * 64 + 32);
    #pragma unroll
    for (int ks = 0; ks < 2; ++ks) {
      v16bf a = ks ? a1 : a0;
      #pragma unroll
      for (int t = 0; t < 4; ++t) {
        v16bf bfrag;
        FRAG_FROM_LDS(bfrag, wbuf[buf], lbase + (unsigned)((t * 16) * LDSROW + ks * 32));
        acc[t] = wmma_bf16(a, bfrag, acc[t]);
      }
    }
    __syncthreads();
  }
  #pragma unroll
  for (int r = 0; r < 8; ++r) {
    size_t row = (size_t)(m0 + r + 8 * lg);
    #pragma unroll
    for (int t = 0; t < 4; ++t) {
      int n = n0 + t * 16 + cl;
      out[row * CD + n] = x[row * CD + n] + acc[t][r] + ob[n];
    }
  }
}

// ---------------------------- host launcher ----------------------------
extern "C" void kernel_launch(void* const* d_in, const int* in_sizes, int n_in,
                              void* d_out, int out_size, void* d_ws, size_t ws_size,
                              hipStream_t stream) {
  const float* x      = (const float*)d_in[0];
  const float* emb    = (const float*)d_in[1];
  const float* mask   = (const float*)d_in[2];
  const float* gamma  = (const float*)d_in[3];
  const float* beta   = (const float*)d_in[4];
  const float* Wq     = (const float*)d_in[5];
  const float* bq     = (const float*)d_in[6];
  const float* Wk     = (const float*)d_in[7];
  const float* bk     = (const float*)d_in[8];
  const float* Wv     = (const float*)d_in[9];
  const float* bv     = (const float*)d_in[10];
  const float* emb_W  = (const float*)d_in[11];
  const float* emb_b  = (const float*)d_in[12];
  const float* gamma2 = (const float*)d_in[13];
  const float* beta2  = (const float*)d_in[14];
  const float* out_W  = (const float*)d_in[15];
  const float* out_b  = (const float*)d_in[16];
  float* out = (float*)d_out;

  char* w = (char*)d_ws;
  size_t off = 0;
  bf16*  xn  = (bf16*)(w + off);  off += (size_t)CM * CD * 2;
  bf16*  wt  = (bf16*)(w + off);  off += (size_t)4 * CD * CD * 2;
  bf16*  qb  = (bf16*)(w + off);  off += (size_t)CM * CD * 2;
  float* kf  = (float*)(w + off); off += (size_t)CM * CD * 4;   // reused as y
  float* vf  = (float*)(w + off); off += (size_t)CM * CD * 4;   // reused as h
  bf16*  at  = (bf16*)(w + off);  off += (size_t)CB * CH * 64 * 64 * 2;
  float* eo  = (float*)(w + off); off += (size_t)CB * 1024 * 4;
  float* pm  = (float*)(w + off); off += (size_t)CB * 4 * 16 * 128 * 4;
  float* ps  = (float*)(w + off); off += (size_t)CB * 4 * 16 * 128 * 4;
  float* fm  = (float*)(w + off); off += (size_t)CB * CD * 4;
  float* fs  = (float*)(w + off); off += (size_t)CB * CD * 4;
  float* yf  = kf;
  bf16*  hb  = (bf16*)vf;

  bf16* wot = wt + (size_t)3 * CD * CD;

  wconv_kernel<<<4096, 256, 0, stream>>>(Wq, Wk, Wv, out_W, wt);
  ln1_kernel<<<CM, 256, 0, stream>>>(x, gamma, beta, xn);
  qkv_gemm_kernel<<<dim3(8, CM / 128), 256, 0, stream>>>(
      xn, wt, bq, bk, bv, mask, qb, kf, vf);
  ksm_partial_kernel<<<dim3(16, 16), 128, 0, stream>>>(kf, pm, ps);
  ksm_combine_kernel<<<16, 128, 0, stream>>>(pm, ps, fm, fs);
  ksm_norm_kernel<<<dim3(16, 16), 128, 0, stream>>>(kf, fm, fs);
  attn_gemm_kernel<<<CB * CH, 256, 0, stream>>>(kf, vf, at);
  y_gemm_kernel<<<dim3(CB * CH, CT / 128), 256, 0, stream>>>(qb, at, yf);
  emb_kernel<<<dim3(CB, 4), 256, 0, stream>>>(emb, emb_W, emb_b, eo);
  ln2mod_kernel<<<CM, 256, 0, stream>>>(yf, gamma2, beta2, eo, hb);
  out_gemm_kernel<<<dim3(8, CM / 128), 256, 0, stream>>>(hb, wot, out_b, x, out);
}
